// EquiLinearLayer_55130200212211
// MI455X (gfx1250) — compile-verified, matrix operand order checked by reading, not run on hardware
//
#include <hip/hip_runtime.h>

// EquiLinear (PGA) layer as a 16384x8192x8192 GEMM:
//   out[(s,n),(j,y)] = sum_{i,x} Bt[(j,y),(i,x)] * x[(s,n),(i,x)]
//   Bt[(j*16+y)][(i*16+x)] = sum_b w[j,i,b] * blade[b,x,y]   (precomputed, bf16)
// Phase 2 is a bf16-WMMA GEMM with f32 accumulation (V_WMMA_F32_16X16X32_BF16).
// Workspace requirement: 8192*8192*2 bytes = 128 MiB in d_ws (fits resident in 192 MB L2).
// x / Out are streamed with non-temporal hints so Bt stays hot in L2.

typedef __attribute__((ext_vector_type(16))) __bf16       v16bf;
typedef __attribute__((ext_vector_type(2)))  __bf16       bf16x2;
typedef __attribute__((ext_vector_type(8)))  float        v8f;
typedef __attribute__((ext_vector_type(4)))  float        f32x4;
typedef __attribute__((ext_vector_type(4)))  unsigned int u32x4;

#define J_DIM 512
#define I_DIM 512
#define BLEN  9
#define M_DIM 16384   // 8 * 2048
#define N_DIM 8192    // 512 * 16
#define K_DIM 8192    // 512 * 16

#define BM 128
#define BN 128
#define BK 32
#define SA_U 20                  // LDS row stride in uints (40 bf16 = 80 B, 16B-aligned, conflict-free)
#define KTILES (K_DIM / BK)      // 256

// f32x2 -> packed bf16x2 via native converts (v_cvt_*bf16*), RNE
__device__ __forceinline__ unsigned int pack2_bf16(float a, float b) {
  bf16x2 h;
  h[0] = (__bf16)a;
  h[1] = (__bf16)b;
  return __builtin_bit_cast(unsigned int, h);
}

// ---------------- Phase 1: Bt[(j*16+y)][(i*16+x)] = sum_b w[j,i,b]*blade[b,x,y] (bf16) ----
__global__ __launch_bounds__(256) void build_Bt_bf16(
    const float* __restrict__ w, const float* __restrict__ blade,
    unsigned short* __restrict__ Bt) {
  __shared__ float bl[BLEN * 256];
  const int t = threadIdx.x;
  for (int r = t; r < BLEN * 256; r += 256) bl[r] = blade[r];
  __syncthreads();

  const int j = blockIdx.x >> 9;     // 0..511
  const int i = blockIdx.x & 511;    // 0..511
  const int y = t >> 4;              // 0..15
  const int x = t & 15;              // 0..15 (consecutive threads -> consecutive k)

  const float* wp = w + ((size_t)j * I_DIM + i) * BLEN;   // uniform -> s_load
  float acc = 0.0f;
#pragma unroll
  for (int b = 0; b < BLEN; ++b)
    acc = fmaf(wp[b], bl[b * 256 + x * 16 + y], acc);

  __bf16 h = (__bf16)acc;
  Bt[(size_t)(j * 16 + y) * K_DIM + i * 16 + x] =
      __builtin_bit_cast(unsigned short, h);
}

// ---------------- Phase 2: C[M,N] = A[M,K] * Bt[N,K]^T  (bf16 WMMA, f32 accum) -----------
union FragBf { u32x4 q[2]; v16bf v; };

__global__ __launch_bounds__(256) void gemm_xBt_wmma(
    const float* __restrict__ X, const unsigned short* __restrict__ Bt,
    float* __restrict__ Out) {
  __shared__ unsigned int As[2][BM * SA_U];   // bf16 tiles, [row][k], padded stride
  __shared__ unsigned int Bs[2][BN * SA_U];   // Bt rows are the GEMM N dim

  const int tid    = threadIdx.x;
  const int lane   = tid & 31;
  const int wid    = tid >> 5;
  const int wave_m = wid & 3;   // 4 waves along M (32 rows each)
  const int wave_n = wid >> 2;  // 2 waves along N (64 cols each)

  const int m0 = blockIdx.y * BM;
  const int n0 = blockIdx.x * BN;

  // ---- cooperative tile loaders: each thread owns 1 row-half (64B of A, 32B of B) ----
  const int lrow  = tid >> 1;   // 0..127
  const int lhalf = tid & 1;    // 0..1  (k sub-chunk of 16 elements)
  const float*          xg = X  + (size_t)(m0 + lrow) * K_DIM + lhalf * 16;
  const unsigned short* bg = Bt + (size_t)(n0 + lrow) * K_DIM + lhalf * 16;
  const int lds_w = lrow * SA_U + lhalf * 8;  // uint offset, 16B aligned

  v8f acc[2][4];
#pragma unroll
  for (int mt = 0; mt < 2; ++mt)
#pragma unroll
    for (int nt = 0; nt < 4; ++nt) {
      v8f z = {0.f, 0.f, 0.f, 0.f, 0.f, 0.f, 0.f, 0.f};
      acc[mt][nt] = z;
    }

  f32x4 aReg[4];
  u32x4 bReg[2];

  // prologue: global loads for k-tile 0, deposit into LDS buffer 0
#pragma unroll
  for (int q = 0; q < 4; ++q)
    aReg[q] = __builtin_nontemporal_load((const f32x4*)(xg + q * 4));
#pragma unroll
  for (int q = 0; q < 2; ++q)
    bReg[q] = *(const u32x4*)(bg + q * 8);
  {
    u32x4 w0, w1;
    w0[0] = pack2_bf16(aReg[0][0], aReg[0][1]);
    w0[1] = pack2_bf16(aReg[0][2], aReg[0][3]);
    w0[2] = pack2_bf16(aReg[1][0], aReg[1][1]);
    w0[3] = pack2_bf16(aReg[1][2], aReg[1][3]);
    w1[0] = pack2_bf16(aReg[2][0], aReg[2][1]);
    w1[1] = pack2_bf16(aReg[2][2], aReg[2][3]);
    w1[2] = pack2_bf16(aReg[3][0], aReg[3][1]);
    w1[3] = pack2_bf16(aReg[3][2], aReg[3][3]);
    *(u32x4*)&As[0][lds_w]     = w0;
    *(u32x4*)&As[0][lds_w + 4] = w1;
    *(u32x4*)&Bs[0][lds_w]     = bReg[0];
    *(u32x4*)&Bs[0][lds_w + 4] = bReg[1];
  }

  // fragment addressing per ISA 16-bit A layout:
  // lane<16: row=lane, K chunks {0..7,16..23}; lane>=16: row=lane-16, K {8..15,24..31}
  const int fr_a   = wave_m * 32 + (lane & 15);
  const int fr_b   = wave_n * 64 + (lane & 15);
  const int fchunk = (lane >> 4) << 2;   // uint offset of first 16B chunk

  for (int kt = 0; kt < KTILES; ++kt) {
    __syncthreads();
    const int  cur  = kt & 1;
    const bool more = (kt + 1) < KTILES;

    if (more) {  // prefetch next k-tile from global while computing
      const int koff = (kt + 1) * BK;
#pragma unroll
      for (int q = 0; q < 4; ++q)
        aReg[q] = __builtin_nontemporal_load((const f32x4*)(xg + koff + q * 4));
#pragma unroll
      for (int q = 0; q < 2; ++q)
        bReg[q] = *(const u32x4*)(bg + koff + q * 8);
    }

    FragBf fa[2], fb[4];
#pragma unroll
    for (int mt = 0; mt < 2; ++mt) {
      const unsigned int* p = &As[cur][(fr_a + mt * 16) * SA_U + fchunk];
      fa[mt].q[0] = *(const u32x4*)(p);
      fa[mt].q[1] = *(const u32x4*)(p + 8);
    }
#pragma unroll
    for (int nt = 0; nt < 4; ++nt) {
      const unsigned int* p = &Bs[cur][(fr_b + nt * 16) * SA_U + fchunk];
      fb[nt].q[0] = *(const u32x4*)(p);
      fb[nt].q[1] = *(const u32x4*)(p + 8);
    }

#pragma unroll
    for (int mt = 0; mt < 2; ++mt)
#pragma unroll
      for (int nt = 0; nt < 4; ++nt)
        acc[mt][nt] = __builtin_amdgcn_wmma_f32_16x16x32_bf16(
            false, fa[mt].v, false, fb[nt].v, (short)0, acc[mt][nt],
            false, false);

    if (more) {  // stage prefetched tile into the other LDS buffer
      const int nxt = cur ^ 1;
      u32x4 w0, w1;
      w0[0] = pack2_bf16(aReg[0][0], aReg[0][1]);
      w0[1] = pack2_bf16(aReg[0][2], aReg[0][3]);
      w0[2] = pack2_bf16(aReg[1][0], aReg[1][1]);
      w0[3] = pack2_bf16(aReg[1][2], aReg[1][3]);
      w1[0] = pack2_bf16(aReg[2][0], aReg[2][1]);
      w1[1] = pack2_bf16(aReg[2][2], aReg[2][3]);
      w1[2] = pack2_bf16(aReg[3][0], aReg[3][1]);
      w1[3] = pack2_bf16(aReg[3][2], aReg[3][3]);
      *(u32x4*)&As[nxt][lds_w]     = w0;
      *(u32x4*)&As[nxt][lds_w + 4] = w1;
      *(u32x4*)&Bs[nxt][lds_w]     = bReg[0];
      *(u32x4*)&Bs[nxt][lds_w + 4] = bReg[1];
    }
  }

  // epilogue: C/D layout (ISA): VGPR v -> row offset v; lanes 16-31 -> row +8; lane%16 -> col
  float* op = Out + (size_t)m0 * N_DIM + n0;
  const int rhalf = (lane >> 4) << 3;
  const int cidx  = lane & 15;
#pragma unroll
  for (int mt = 0; mt < 2; ++mt) {
#pragma unroll
    for (int nt = 0; nt < 4; ++nt) {
      const int rb = wave_m * 32 + mt * 16 + rhalf;
      const int cb = wave_n * 64 + nt * 16 + cidx;
#pragma unroll
      for (int v = 0; v < 8; ++v)
        __builtin_nontemporal_store(acc[mt][nt][v],
                                    op + (size_t)(rb + v) * N_DIM + cb);
    }
  }
}

extern "C" void kernel_launch(void* const* d_in, const int* in_sizes, int n_in,
                              void* d_out, int out_size, void* d_ws, size_t ws_size,
                              hipStream_t stream) {
  (void)in_sizes; (void)n_in; (void)out_size; (void)ws_size;
  const float* x     = (const float*)d_in[0];  // [8,2048,512,16] f32
  const float* w     = (const float*)d_in[1];  // [512,512,9]     f32
  const float* blade = (const float*)d_in[2];  // [9,16,16]       f32
  float* out = (float*)d_out;                  // [8,2048,512,16] f32
  unsigned short* Bt = (unsigned short*)d_ws;  // 8192*8192 bf16 = 128 MiB scratch

  build_Bt_bf16<<<dim3(J_DIM * I_DIM), 256, 0, stream>>>(w, blade, Bt);
  gemm_xBt_wmma<<<dim3(N_DIM / BN, M_DIM / BM), 256, 0, stream>>>(x, Bt, out);
}
